// InvariantAttention_73564199846148
// MI455X (gfx1250) — matrix operand order repaired and checked by reading, not compiled, not run on hardware
//
#include <hip/hip_runtime.h>
#include <hip/hip_bf16.h>

typedef __attribute__((ext_vector_type(16))) _Float16 v16h;
typedef __attribute__((ext_vector_type(8)))  float    v8f;

union H16 { v16h v; _Float16 e[16]; };
union F8  { v8f  v; float    e[8];  };

constexpr int B = 4, T = 512, E = 512, H = 512, NH = 16, D = 32, SRC = T + E;
constexpr float SCALING = 0.17677669529663687f;   // sqrt(32)/32
constexpr float LN_EPS  = 1e-5f;

__device__ __forceinline__ v8f wmma16(v16h a, v16h b, v8f c) {
  // D = A(16x32 f16) x B(32x16 f16) + C(16x16 f32)
  return __builtin_amdgcn_wmma_f32_16x16x32_f16(false, a, false, b, (short)0, c, false, false);
}

__device__ __forceinline__ int a_kidx(int j, int hh) {
  // A-matrix 16x32 layout (contiguous, matches B fills): lanes 0-15 K={0..15}, lanes 16-31 K={16..31}
  return hh * 16 + 2 * j;
}

// ---- DPP16 16-lane all-reduce (pure VALU, no LDS, no dscnt waits) ----
template <int CTRL>
__device__ __forceinline__ float dpp_mov_f32(float x) {
  int xi = __float_as_int(x);
  int yi = __builtin_amdgcn_update_dpp(xi, xi, CTRL, 0xF, 0xF, true);
  return __int_as_float(yi);
}
__device__ __forceinline__ float rowmax16(float x) {
  x = fmaxf(x, dpp_mov_f32<0xB1>(x));    // quad_perm [1,0,3,2] : xor 1
  x = fmaxf(x, dpp_mov_f32<0x4E>(x));    // quad_perm [2,3,0,1] : xor 2
  x = fmaxf(x, dpp_mov_f32<0x141>(x));   // row_half_mirror     : xor 7 (crosses quads)
  x = fmaxf(x, dpp_mov_f32<0x140>(x));   // row_mirror          : xor 15 (crosses 8-groups)
  return x;
}
__device__ __forceinline__ float rowsum16(float x) {
  x += dpp_mov_f32<0xB1>(x);
  x += dpp_mov_f32<0x4E>(x);
  x += dpp_mov_f32<0x141>(x);
  x += dpp_mov_f32<0x140>(x);
  return x;
}

// ---------------- prep kernels ----------------
__global__ void prep_scale_q(const float* __restrict__ q, _Float16* __restrict__ qf) {
  int i = blockIdx.x * blockDim.x + threadIdx.x;
  if (i < B * T * H) qf[i] = (_Float16)(q[i] * SCALING);
}

__global__ void prep_w(const float* __restrict__ w, _Float16* __restrict__ wf) {
  int i = blockIdx.x * blockDim.x + threadIdx.x;
  if (i < H * H) wf[i] = (_Float16)w[i];
}

__global__ void prep_kexp(const float* __restrict__ k, const long long* __restrict__ idx,
                          _Float16* __restrict__ kexp) {
  int i = blockIdx.x * blockDim.x + threadIdx.x;
  if (i >= B * SRC * H) return;
  int hc = i % H;
  int s  = (i / H) % SRC;
  int b  = i / (H * SRC);
  int ts = (s < T) ? s : (int)idx[b * E + (s - T)];
  kexp[i] = (_Float16)k[((size_t)b * T + ts) * H + hc];
}

__global__ void prep_vt(const float* __restrict__ v, const long long* __restrict__ idx,
                        _Float16* __restrict__ vt) {
  // vt layout: [b][nh][d][s]  (s contiguous -> PV B-tile dword loads)
  int i = blockIdx.x * blockDim.x + threadIdx.x;
  if (i >= B * NH * D * SRC) return;
  int s  = i % SRC;
  int d  = (i / SRC) % D;
  int nh = (i / (SRC * D)) % NH;
  int b  = i / (SRC * D * NH);
  int ts = (s < T) ? s : (int)idx[b * E + (s - T)];
  vt[i] = (_Float16)v[((size_t)b * T + ts) * H + nh * D + d];
}

// ---------------- fused attention (flash-style, online renorm, branchless masks,
// ---------------- DPP row-max, per-lane deferred denominator) ----------------
__global__ __launch_bounds__(256) void attn_fused(
    const _Float16* __restrict__ qf, const _Float16* __restrict__ kexp,
    const _Float16* __restrict__ vt, const float* __restrict__ bias,
    const float* __restrict__ lmask, const unsigned char* __restrict__ emask,
    float* __restrict__ attn)
{
  __shared__ _Float16 lds_p[8][16][32];   // per-wave P staging (C-layout -> A-layout)
  const int wid  = threadIdx.x >> 5;
  const int lane = threadIdx.x & 31;
  const int hh   = lane >> 4;             // wave half
  const int ln   = lane & 15;
  const int b    = blockIdx.x / NH;
  const int h    = blockIdx.x % NH;
  const int t0   = blockIdx.y * 128 + wid * 16;

  // Q A-tile (16 rows x 32 d), scaled already
  H16 aq;
  {
    const _Float16* qrow = qf + ((size_t)(b * T + t0 + ln)) * H + h * D;
    #pragma unroll
    for (int j = 0; j < 8; ++j) {
      const int kd = a_kidx(j, hh);
      aq.e[2 * j]     = qrow[kd];
      aq.e[2 * j + 1] = qrow[kd + 1];
    }
  }

  float Mrow[8], Lrow[8];   // Lrow: PER-LANE partial denominator (reduced once at the end)
  #pragma unroll
  for (int r = 0; r < 8; ++r) { Mrow[r] = -__builtin_inff(); Lrow[r] = 0.f; }
  F8 o0 = {}, o1 = {};   // PV accumulators: d = 0..15 and 16..31

  const float* bias_bh = bias + ((size_t)(b * NH + h)) * T * SRC;
  const float* lm_b    = lmask + ((size_t)b * T) * SRC;
  const unsigned char* em_b = emask + (size_t)b * SRC;
  const _Float16* vbase = vt + ((size_t)(b * NH + h)) * D * SRC;

  for (int ic = 0; ic < SRC / 32; ++ic) {
    const int s0 = ic * 32;

    // K^T B-tiles: col n = s, rows = d (contiguous pairs in kexp row)
    H16 bk0, bk1;
    {
      const _Float16* kr0 = kexp + ((size_t)b * SRC + s0 + ln) * H + h * D + hh * 16;
      const _Float16* kr1 = kr0 + (size_t)16 * H;
      #pragma unroll
      for (int j = 0; j < 8; ++j) {
        bk0.e[2 * j] = kr0[2 * j]; bk0.e[2 * j + 1] = kr0[2 * j + 1];
        bk1.e[2 * j] = kr1[2 * j]; bk1.e[2 * j + 1] = kr1[2 * j + 1];
      }
    }
    F8 c0 = {}, c1 = {};
    c0.v = wmma16(aq.v, bk0.v, c0.v);
    c1.v = wmma16(aq.v, bk1.v, c1.v);

    if (ic + 1 < SRC / 32)  // speculative prefetch of next bias chunk
      __builtin_prefetch(bias_bh + (size_t)(t0 + (lane & 15)) * SRC + s0 + 32, 0, 0);

    const int sa = s0 + ln, sb = s0 + 16 + ln;
    // column validity as float multipliers (branchless)
    const float emv0 = (em_b[sa] != 0) ? 0.f : 1.f;
    const float emv1 = (em_b[sb] != 0) ? 0.f : 1.f;

    // Unconditional loads + logits. Masked entries keep their (finite) logit:
    // including them in the row max is harmless (max only needs to upper-bound
    // the valid logits; the shift cancels in the normalization), and their
    // weight is zeroed via vm = masked ? 0 : local_mask.
    float xv0[8], xv1[8], l0a[8], l1a[8], vm0[8], vm1[8];
    #pragma unroll
    for (int r = 0; r < 8; ++r) {
      const int t = t0 + r + 8 * hh;      // C-layout row for this lane half
      const float b0 = bias_bh[(size_t)t * SRC + sa];
      const float b1 = bias_bh[(size_t)t * SRC + sb];
      const float l0 = lm_b[(size_t)t * SRC + sa];
      const float l1 = lm_b[(size_t)t * SRC + sb];
      l0a[r] = l0; l1a[r] = l1;
      vm0[r] = (l0 > 1e-6f ? l0 : 0.f) * emv0;
      vm1[r] = (l1 > 1e-6f ? l1 : 0.f) * emv1;
      xv0[r] = c0.e[r] + b0;
      xv1[r] = c1.e[r] + b1;
    }

    #pragma unroll
    for (int r = 0; r < 8; ++r) {
      // Row max must be cross-lane and row-uniform (feeds the rescale factor
      // and the f16-staged P values); DPP16 butterfly keeps it VALU-only.
      const float cm = rowmax16(fmaxf(xv0[r], xv1[r]));
      const float newM = fmaxf(Mrow[r], cm);          // finite after first chunk
      const float scale = __expf(Mrow[r] - newM);     // first chunk: exp(-inf)=0, acc is 0 anyway
      const float w0 = __expf(xv0[r] - newM) * vm0[r];   // e * m  (denominator weight)
      const float w1 = __expf(xv1[r] - newM) * vm1[r];
      Lrow[r] = Lrow[r] * scale + (w0 + w1);          // per-lane partial (scale is row-uniform)
      Mrow[r] = newM;
      o0.e[r] *= scale;
      o1.e[r] *= scale;
      const int mr = r + 8 * hh;
      lds_p[wid][mr][ln]      = (_Float16)(w0 * l0a[r]);   // e * m^2 (numerator weight)
      lds_p[wid][mr][16 + ln] = (_Float16)(w1 * l1a[r]);
    }
    __syncthreads();

    // P as A-tile (16 x 32) from LDS
    H16 ap;
    #pragma unroll
    for (int j = 0; j < 8; ++j) {
      const int kd = a_kidx(j, hh);
      ap.e[2 * j]     = lds_p[wid][ln][kd];
      ap.e[2 * j + 1] = lds_p[wid][ln][kd + 1];
    }
    // V B-tiles: col n = d, rows = s (contiguous pairs in vt)
    H16 bv0, bv1;
    {
      const _Float16* vr0 = vbase + (size_t)ln * SRC + s0 + hh * 16;
      const _Float16* vr1 = vbase + (size_t)(16 + ln) * SRC + s0 + hh * 16;
      #pragma unroll
      for (int j = 0; j < 8; ++j) {
        bv0.e[2 * j] = vr0[2 * j]; bv0.e[2 * j + 1] = vr0[2 * j + 1];
        bv1.e[2 * j] = vr1[2 * j]; bv1.e[2 * j + 1] = vr1[2 * j + 1];
      }
    }
    o0.v = wmma16(ap.v, bv0.v, o0.v);
    o1.v = wmma16(ap.v, bv1.v, o1.v);
    __syncthreads();
  }

  // One final cross-lane reduction of the per-lane denominator partials.
  #pragma unroll
  for (int r = 0; r < 8; ++r) Lrow[r] = rowsum16(Lrow[r]);

  float* arow = attn + ((size_t)b * T) * H + (size_t)h * D;
  #pragma unroll
  for (int r = 0; r < 8; ++r) {
    const int t = t0 + r + 8 * hh;
    const float inv = 1.f / (Lrow[r] + 1e-10f);
    arow[(size_t)t * H + ln]      = o0.e[r] * inv;
    arow[(size_t)t * H + 16 + ln] = o1.e[r] * inv;
  }
}

// ---------------- LayerNorm (one wave per row) ----------------
__global__ __launch_bounds__(256) void ln_rows(const float* __restrict__ attn,
    const float* __restrict__ g, const float* __restrict__ be, _Float16* __restrict__ xln)
{
  const int wid = threadIdx.x >> 5, lane = threadIdx.x & 31;
  const int row = blockIdx.x * 8 + wid;
  const float* x = attn + (size_t)row * H;
  float vals[16]; float s = 0.f;
  #pragma unroll
  for (int i = 0; i < 16; ++i) { vals[i] = x[lane + 32 * i]; s += vals[i]; }
  #pragma unroll
  for (int m = 1; m < 32; m <<= 1) s += __shfl_xor(s, m, 32);
  const float mu = s * (1.f / H);
  float v2 = 0.f;
  #pragma unroll
  for (int i = 0; i < 16; ++i) { const float d = vals[i] - mu; v2 += d * d; }
  #pragma unroll
  for (int m = 1; m < 32; m <<= 1) v2 += __shfl_xor(v2, m, 32);
  const float inv = rsqrtf(v2 * (1.f / H) + LN_EPS);
  #pragma unroll
  for (int i = 0; i < 16; ++i) {
    const int c = lane + 32 * i;
    xln[(size_t)row * H + c] = (_Float16)((vals[i] - mu) * inv * g[c] + be[c]);
  }
}

// ---------------- output projection via WMMA ----------------
__global__ __launch_bounds__(256) void proj_wmma(const _Float16* __restrict__ xln,
    const _Float16* __restrict__ wf, const float* __restrict__ ob, float* __restrict__ out)
{
  const int wid = threadIdx.x >> 5, lane = threadIdx.x & 31;
  const int hh = lane >> 4, ln = lane & 15;
  const int t0  = blockIdx.x * 32 + (wid >> 2) * 16;
  const int oc0 = blockIdx.y * 64 + (wid & 3) * 16;
  F8 c = {};
  for (int kc = 0; kc < H / 32; ++kc) {
    const int kb = kc * 32;
    H16 a, w;
    const _Float16* ar = xln + (size_t)(t0 + ln) * H + kb;
    const _Float16* wr = wf + (size_t)(oc0 + ln) * H + kb + hh * 16;  // B: out = xln @ out_w^T
    #pragma unroll
    for (int j = 0; j < 8; ++j) {
      const int kd = a_kidx(j, hh);
      a.e[2 * j] = ar[kd];     a.e[2 * j + 1] = ar[kd + 1];
      w.e[2 * j] = wr[2 * j];  w.e[2 * j + 1] = wr[2 * j + 1];
    }
    c.v = wmma16(a.v, w.v, c.v);
  }
  const float bias = ob[oc0 + ln];
  #pragma unroll
  for (int r = 0; r < 8; ++r)
    out[(size_t)(t0 + r + 8 * hh) * H + oc0 + ln] = c.e[r] + bias;
}

// ---------------- launcher ----------------
extern "C" void kernel_launch(void* const* d_in, const int* in_sizes, int n_in,
                              void* d_out, int out_size, void* d_ws, size_t ws_size,
                              hipStream_t stream) {
  const float* q  = (const float*)d_in[0];
  const float* k  = (const float*)d_in[1];
  const float* v  = (const float*)d_in[2];
  const float* ab = (const float*)d_in[3];
  const float* lm = (const float*)d_in[4];
  const unsigned char* em = (const unsigned char*)d_in[5];   // bool array
  const long long* idx = (const long long*)d_in[6];          // int64 (x64 enabled)
  const float* ow  = (const float*)d_in[7];
  const float* obv = (const float*)d_in[8];
  const float* lg  = (const float*)d_in[9];
  const float* lb  = (const float*)d_in[10];
  float* out = (float*)d_out;

  char* ws = (char*)d_ws;
  _Float16* qf   = (_Float16*)(ws + 0);           //  2 MB  (B*T*H f16)
  _Float16* kex  = (_Float16*)(ws + 2097152);     //  4 MB  (B*SRC*H f16)
  _Float16* vtp  = (_Float16*)(ws + 6291456);     //  4 MB  (B*NH*D*SRC f16)
  float*    attn = (float*)   (ws + 10485760);    //  4 MB  (B*T*H f32)
  _Float16* xln  = (_Float16*)(ws + 14680064);    //  2 MB  (B*T*H f16)
  _Float16* wf   = (_Float16*)(ws + 16777216);    // 0.5 MB (H*H f16)

  prep_scale_q<<<(B * T * H + 255) / 256, 256, 0, stream>>>(q, qf);
  prep_w<<<(H * H + 255) / 256, 256, 0, stream>>>(ow, wf);
  prep_kexp<<<(B * SRC * H + 255) / 256, 256, 0, stream>>>(k, idx, kex);
  prep_vt<<<(B * NH * D * SRC + 255) / 256, 256, 0, stream>>>(v, idx, vtp);

  attn_fused<<<dim3(B * NH, T / 128), 256, 0, stream>>>(qf, kex, vtp, ab, lm, em, attn);

  ln_rows<<<(B * T) / 8, 256, 0, stream>>>(attn, lg, lb, xln);
  proj_wmma<<<dim3((B * T) / 32, H / 64), 256, 0, stream>>>(xln, wf, obv, out);
}
